// VectorQuantizer_68444598829402
// MI455X (gfx1250) — matrix operand order repaired and checked by reading, not compiled, not run on hardware
//
#include <hip/hip_runtime.h>
#include <hip/hip_bf16.h>

typedef __bf16 v16bf __attribute__((ext_vector_type(16)));
typedef float  v8f   __attribute__((ext_vector_type(8)));

#define N_E      2048
#define E_DIM    128
#define M_TOTAL  32768
#define BETA     0.25f

// d_out layout (flat floats, reference tuple order)
#define OFF_LOSS ((size_t)0)
#define OFF_ZQ   ((size_t)1)
#define OFF_PERP ((size_t)(1 + (size_t)M_TOTAL * E_DIM))            // 4194305
#define OFF_ME   (OFF_PERP + 1)                                     // 4194306
#define OFF_IDX  (OFF_ME + (size_t)M_TOTAL * N_E)                   // 71303170

// ---------------------------------------------------------------------------
// Kernel 1: codebook norms + zero histogram/SSE scratch (every call: replay-safe)
// ---------------------------------------------------------------------------
__global__ void vq_prep(const float* __restrict__ emb, float* __restrict__ norms,
                        int* __restrict__ counts, float* __restrict__ sse) {
  int t = blockIdx.x * blockDim.x + threadIdx.x;
  if (t < N_E) {
    const float* e = emb + (size_t)t * E_DIM;
    float s = 0.f;
    #pragma unroll 4
    for (int k = 0; k < E_DIM; ++k) s += e[k] * e[k];
    norms[t]  = s;
    counts[t] = 0;
  }
  if (t == 0) *sse = 0.f;
}

// ---------------------------------------------------------------------------
// Kernel 2: fused distance-GEMM (bf16 hi/lo split WMMA) + argmin + gather
// 256 blocks x 256 threads (8 waves). Wave w owns rows blk*128 + w*16 .. +15.
// ---------------------------------------------------------------------------
__launch_bounds__(256)
__global__ void vq_main(const float* __restrict__ z, const float* __restrict__ emb,
                        const float* __restrict__ norms, int* __restrict__ counts,
                        float* __restrict__ sse_acc, float* __restrict__ out) {
  // emb tile: 64 rows x 128 K, hi/lo bf16, pitch 136 (272B: 16B-aligned, bank-safe)
  __shared__ __align__(16) __bf16 sBhi[64 * 136];
  __shared__ __align__(16) __bf16 sBlo[64 * 136];
  __shared__ float sNorm[64];

  const int tid  = threadIdx.x;
  const int wave = tid >> 5;
  const int lane = tid & 31;
  const int lm   = lane & 15;    // row (A) / col (B,C) within 16
  const int h    = lane >> 4;    // lane-half selects K phase
  const int mbase = blockIdx.x * 128 + wave * 16;

  // ---- Preload A fragments (z rows) as bf16 hi/lo, reused for all 2048 cols.
  // 16-bit A 16x32 layout (contiguous, matches B): elems 0..7 -> K = 32c + 16h + e,
  //                            elems 8..15 -> K = 32c + 16h + 8 + e.
  v16bf Ahi[4], Alo[4];
  {
    const float* zrow = z + (size_t)(mbase + lm) * E_DIM;
    #pragma unroll
    for (int c = 0; c < 4; ++c) {
      const int k0 = c * 32 + 16 * h;
      #pragma unroll
      for (int e = 0; e < 8; ++e) {
        float f0 = zrow[k0 + e];
        float f1 = zrow[k0 + 8 + e];
        __bf16 h0 = (__bf16)f0;
        __bf16 h1 = (__bf16)f1;
        Ahi[c][e]     = h0;
        Ahi[c][8 + e] = h1;
        Alo[c][e]     = (__bf16)(f0 - (float)h0);
        Alo[c][8 + e] = (__bf16)(f1 - (float)h1);
      }
    }
  }

  float minv[8];
  int   mini[8];
  #pragma unroll
  for (int j = 0; j < 8; ++j) { minv[j] = 3.4e38f; mini[j] = 0; }

  for (int nt = 0; nt < N_E / 64; ++nt) {
    const int nbase = nt * 64;
    __syncthreads();
    // ---- Stage emb tile (64x128 fp32 -> hi/lo bf16) cooperatively.
    for (int i = tid; i < 64 * 32; i += 256) {       // 2048 float4 chunks
      int row = i >> 5;
      int c4  = (i & 31) * 4;
      float4 f = *reinterpret_cast<const float4*>(emb + (size_t)(nbase + row) * E_DIM + c4);
      __bf16* ph = &sBhi[row * 136 + c4];
      __bf16* pl = &sBlo[row * 136 + c4];
      __bf16 hx = (__bf16)f.x; ph[0] = hx; pl[0] = (__bf16)(f.x - (float)hx);
      __bf16 hy = (__bf16)f.y; ph[1] = hy; pl[1] = (__bf16)(f.y - (float)hy);
      __bf16 hz = (__bf16)f.z; ph[2] = hz; pl[2] = (__bf16)(f.z - (float)hz);
      __bf16 hw = (__bf16)f.w; ph[3] = hw; pl[3] = (__bf16)(f.w - (float)hw);
    }
    if (tid < 64) sNorm[tid] = norms[nbase + tid];
    __syncthreads();

    // ---- 4 sub-tiles of 16 columns each.
    #pragma unroll
    for (int sub = 0; sub < 4; ++sub) {
      const int trow = sub * 16 + lm;                // emb row in tile = C column
      const __bf16* bh = &sBhi[trow * 136];
      const __bf16* bl = &sBlo[trow * 136];
      v8f C = {0.f, 0.f, 0.f, 0.f, 0.f, 0.f, 0.f, 0.f};
      #pragma unroll
      for (int c = 0; c < 4; ++c) {
        // B in contiguous K order (same as A): lane half = K+16, 16 contiguous K per lane.
        v16bf Bh = *reinterpret_cast<const v16bf*>(bh + c * 32 + 16 * h);
        v16bf Bl = *reinterpret_cast<const v16bf*>(bl + c * 32 + 16 * h);
        // hi*hi + hi*lo + lo*hi  ->  ~fp32-accurate z.e, fp32 accumulate
        C = __builtin_amdgcn_wmma_f32_16x16x32_bf16(false, Ahi[c], false, Bh,
                                                    (short)0, C, false, false);
        C = __builtin_amdgcn_wmma_f32_16x16x32_bf16(false, Ahi[c], false, Bl,
                                                    (short)0, C, false, false);
        C = __builtin_amdgcn_wmma_f32_16x16x32_bf16(false, Alo[c], false, Bh,
                                                    (short)0, C, false, false);
      }
      const float nrm  = sNorm[trow];
      const int   ncol = nbase + trow;               // candidate embedding index
      #pragma unroll
      for (int j = 0; j < 8; ++j) {                  // C row = j + 8h
        float d = nrm - 2.0f * C[j];
        if (d < minv[j]) { minv[j] = d; mini[j] = ncol; }  // strict < keeps earliest n
      }
    }
  }

  // ---- Reduce argmin across the 16 lanes of each half (tie -> smaller index).
  #pragma unroll
  for (int j = 0; j < 8; ++j) {
    #pragma unroll
    for (int m = 1; m < 16; m <<= 1) {
      float ov = __shfl_xor(minv[j], m, 32);
      int   oi = __shfl_xor(mini[j], m, 32);
      if (ov < minv[j] || (ov == minv[j] && oi < mini[j])) { minv[j] = ov; mini[j] = oi; }
    }
  }
  int oidx[8];                                       // other half's rows
  #pragma unroll
  for (int j = 0; j < 8; ++j) oidx[j] = __shfl_xor(mini[j], 16, 32);

  // ---- Write idx + histogram (one writer lane per half).
  if (lm == 0) {
    #pragma unroll
    for (int j = 0; j < 8; ++j) {
      int m = mbase + j + 8 * h;
      out[OFF_IDX + m] = (float)mini[j];
      atomicAdd(&counts[mini[j]], 1);
    }
  }

  // ---- Gather z_q = emb[idx], write z_q_st, accumulate SSE.
  float sse = 0.f;
  #pragma unroll
  for (int r = 0; r < 16; ++r) {
    const int j = r & 7, rh = r >> 3;
    const int idx_r = (rh == h) ? mini[j] : oidx[j]; // wave-uniform value
    float4 e4 = *reinterpret_cast<const float4*>(emb + (size_t)idx_r * E_DIM + lane * 4);
    float4 z4 = *reinterpret_cast<const float4*>(z + (size_t)(mbase + r) * E_DIM + lane * 4);
    float* qp = out + OFF_ZQ + (size_t)(mbase + r) * E_DIM + lane * 4; // 4B-aligned section
    qp[0] = e4.x; qp[1] = e4.y; qp[2] = e4.z; qp[3] = e4.w;
    float dx = e4.x - z4.x, dy = e4.y - z4.y, dz = e4.z - z4.z, dw = e4.w - z4.w;
    sse += dx * dx + dy * dy + dz * dz + dw * dw;
  }
  #pragma unroll
  for (int m = 1; m < 32; m <<= 1) sse += __shfl_xor(sse, m, 32);
  if (lane == 0) atomicAdd(sse_acc, sse);
}

// ---------------------------------------------------------------------------
// Kernel 3: min_encodings one-hot (268 MB: zero-fill + scatter 1.0)
// 4096 blocks x 256 threads, 8 rows per block. ME section is 8-mod-16B -> float2.
// ---------------------------------------------------------------------------
__global__ void vq_onehot(const float* __restrict__ idx_f, float* __restrict__ me) {
  const int blk = blockIdx.x;
  const int tid = threadIdx.x;
  #pragma unroll
  for (int r = 0; r < 8; ++r) {
    float2* row = reinterpret_cast<float2*>(me + (size_t)(blk * 8 + r) * N_E);
    #pragma unroll
    for (int i = 0; i < 4; ++i) row[tid + 256 * i] = make_float2(0.f, 0.f);
  }
  __syncthreads();
  if (tid < 8) {
    size_t m = (size_t)blk * 8 + tid;
    int id = (int)idx_f[m];
    me[m * N_E + id] = 1.0f;
  }
}

// ---------------------------------------------------------------------------
// Kernel 4: scalars (loss, perplexity)
// ---------------------------------------------------------------------------
__global__ void vq_finalize(const int* __restrict__ counts, const float* __restrict__ sse,
                            float* __restrict__ out) {
  __shared__ float red[256];
  const int t = threadIdx.x;
  float s = 0.f;
  for (int i = t; i < N_E; i += 256) {
    float p = (float)counts[i] * (1.0f / (float)M_TOTAL);
    s += p * logf(p + 1e-10f);
  }
  red[t] = s;
  __syncthreads();
  for (int w = 128; w > 0; w >>= 1) {
    if (t < w) red[t] += red[t + w];
    __syncthreads();
  }
  if (t == 0) {
    out[OFF_PERP] = expf(-red[0]);
    out[OFF_LOSS] = (*sse) * (1.0f + BETA) / (float)((size_t)M_TOTAL * E_DIM);
  }
}

// ---------------------------------------------------------------------------
extern "C" void kernel_launch(void* const* d_in, const int* in_sizes, int n_in,
                              void* d_out, int out_size, void* d_ws, size_t ws_size,
                              hipStream_t stream) {
  (void)in_sizes; (void)n_in; (void)out_size; (void)ws_size;
  const float* z   = (const float*)d_in[0];   // [16,2048,128] fp32
  const float* emb = (const float*)d_in[1];   // [2048,128] fp32
  float* out = (float*)d_out;

  int*   counts = (int*)d_ws;                               // 2048 ints
  float* sse    = (float*)((char*)d_ws + 2048 * 4);         // 1 float
  float* norms  = (float*)((char*)d_ws + 2048 * 4 + 16);    // 2048 floats

  vq_prep    <<<8,    256, 0, stream>>>(emb, norms, counts, sse);
  vq_main    <<<256,  256, 0, stream>>>(z, emb, norms, counts, sse, out);
  vq_onehot  <<<4096, 256, 0, stream>>>(out + OFF_IDX, out + OFF_ME);
  vq_finalize<<<1,    256, 0, stream>>>(counts, sse, out);
}
